// QWenAttention_6760278524504
// MI455X (gfx1250) — compile-verified, matrix operand order checked by reading, not run on hardware
//
#include <hip/hip_runtime.h>

// ---------------- problem constants ----------------
#define Bc   2
#define Sc   2048
#define Hc   2048
#define NHc  16
#define HDc  128
#define H3c  (3 * Hc)          // 6144
#define BSc  (Bc * Sc)         // 4096
#define NEGF (-1.0e9f)
#define SCALEF 0.08838834764831845f   // 1/sqrt(128)

typedef __attribute__((ext_vector_type(16))) __bf16 v16bf;
typedef __attribute__((ext_vector_type(8)))  float  v8f;

union AFrag {                 // 16x32 bf16 A-operand (or 32x16 B-operand) per ISA layout
    v16bf v;
    unsigned int u[8];
};

__device__ __forceinline__ unsigned short f2b(float f) {
    unsigned int u = __float_as_uint(f);
    unsigned int r = u + 0x7FFFu + ((u >> 16) & 1u);   // round-to-nearest-even
    return (unsigned short)(r >> 16);
}
__device__ __forceinline__ float b2f(unsigned short s) {
    return __uint_as_float(((unsigned int)s) << 16);
}

// K-pair base within a 32-wide K slab, per the CDNA5 16-bit A/B VGPR layout:
// VGPR r (r<4): lanes0-15 K=2r, lanes16-31 K=8+2r ; VGPR r (r>=4): +16.
__device__ __forceinline__ int kpair(int r, int hi) {
    return (r < 4) ? (hi * 8 + 2 * r) : (16 + hi * 8 + 2 * (r - 4));
}

// Load a 16x32 bf16 fragment. For the A operand: row0+(lane&15) selects M row.
// For the B operand (matrix stored transposed, [N][K]): row0+(lane&15) selects N col.
// K pairs are contiguous ushorts -> merges into global_load_b128 clauses.
__device__ __forceinline__ void load_frag(AFrag& f, const unsigned short* base,
                                          int stride, int row0, int kcol, int lane) {
    const int m  = lane & 15;
    const int hi = lane >> 4;
    #pragma unroll
    for (int r = 0; r < 8; ++r) {
        size_t off = (size_t)(row0 + m) * (size_t)stride + (size_t)(kcol + kpair(r, hi));
        f.u[r] = *(const unsigned int*)(base + off);
    }
}

__device__ __forceinline__ v8f wmma_bf16(const AFrag& a, const AFrag& b, v8f c) {
    return __builtin_amdgcn_wmma_f32_16x16x32_bf16(false, a.v, false, b.v,
                                                   (short)0, c, false, false);
}

// ---------------- kernel 1: fp32 -> bf16 copy ----------------
__global__ void cvt_f2b_kernel(const float* __restrict__ src,
                               unsigned short* __restrict__ dst, int n) {
    int i = blockIdx.x * blockDim.x + threadIdx.x;
    if (i < n) dst[i] = f2b(src[i]);
}

// ---------------- kernel 2: fp32 [K][N] -> bf16 [N][K] transpose ----------------
__global__ void transpose_f2b_kernel(const float* __restrict__ W,
                                     unsigned short* __restrict__ Wt, int K, int N) {
    int i = blockIdx.x * blockDim.x + threadIdx.x;
    if (i < K * N) {
        int k = i / N, n = i - k * N;
        Wt[(size_t)n * K + k] = f2b(W[i]);
    }
}

// ---------------- kernels 3/7: WMMA GEMM, C = A[MxK] * Bt[NxK]^T + bias ----------------
// block = 256 threads (8 waves). Wave tile = 32x64 (2 A-frags x 4 B-frags, 8 WMMA/K-step),
// block tile = 64x256.
template <bool OUT_BF16>
__global__ __launch_bounds__(256)
void gemm_wmma_kernel(const unsigned short* __restrict__ A,
                      const unsigned short* __restrict__ Bt,
                      const float* __restrict__ bias,
                      void* __restrict__ Cout, int M, int N, int K) {
    const int wave = threadIdx.x >> 5;
    const int lane = threadIdx.x & 31;
    const int m0 = blockIdx.y * 64 + (wave & 1) * 32;
    const int n0 = blockIdx.x * 256 + (wave >> 1) * 64;
    const int ml = lane & 15;

    v8f acc[2][4];
    #pragma unroll
    for (int a = 0; a < 2; ++a)
        #pragma unroll
        for (int t = 0; t < 4; ++t) acc[a][t] = (v8f)0.0f;

    for (int k0 = 0; k0 < K; k0 += 32) {
        // prefetch K slab two steps ahead (global_prefetch_b8; no counter cost)
        if (k0 + 64 < K) {
            __builtin_prefetch(A + (size_t)(m0 + ml) * K + k0 + 64, 0, 1);
            __builtin_prefetch(A + (size_t)(m0 + 16 + ml) * K + k0 + 64, 0, 1);
            #pragma unroll
            for (int t = 0; t < 4; ++t)
                __builtin_prefetch(Bt + (size_t)(n0 + 16 * t + ml) * K + k0 + 64, 0, 1);
        }
        AFrag a0, a1;
        load_frag(a0, A, K, m0,      k0, lane);
        load_frag(a1, A, K, m0 + 16, k0, lane);
        #pragma unroll
        for (int t = 0; t < 4; ++t) {
            AFrag b;
            load_frag(b, Bt, K, n0 + 16 * t, k0, lane);
            acc[0][t] = wmma_bf16(a0, b, acc[0][t]);
            acc[1][t] = wmma_bf16(a1, b, acc[1][t]);
        }
    }

    const int hi = lane >> 4;
    #pragma unroll
    for (int a = 0; a < 2; ++a) {
        #pragma unroll
        for (int t = 0; t < 4; ++t) {
            #pragma unroll
            for (int r = 0; r < 8; ++r) {
                int row = m0 + 16 * a + r + 8 * hi;
                int col = n0 + 16 * t + ml;
                float val = acc[a][t][r] + bias[col];
                if (OUT_BF16)
                    ((unsigned short*)Cout)[(size_t)row * N + col] = f2b(val);
                else
                    ((float*)Cout)[(size_t)row * N + col] = val;
            }
        }
    }
}

// ---------------- kernel 4: RoPE + logn + layout shuffle ----------------
// qkv bf16 [B,S,3H] -> q,k bf16 [B*NH, S, HD], v bf16 transposed [B*NH, HD, S]
__global__ void rope_kernel(const unsigned short* __restrict__ qkv,
                            const float* __restrict__ cosp,   // [S, HD]
                            const float* __restrict__ sinp,   // [S, HD]
                            const float* __restrict__ logn,   // [S]
                            unsigned short* __restrict__ qb,
                            unsigned short* __restrict__ kb,
                            unsigned short* __restrict__ vtb) {
    int tid = blockIdx.x * blockDim.x + threadIdx.x;
    if (tid >= Bc * NHc * Sc * (HDc / 2)) return;
    int d = tid & 63;  int t = tid >> 6;
    int s = t & (Sc - 1); t >>= 11;
    int h = t & (NHc - 1); int b = t >> 4;

    const size_t base = (size_t)(b * Sc + s) * H3c + h * HDc;
    float q0 = b2f(qkv[base + d]),            q1 = b2f(qkv[base + d + 64]);
    float k0 = b2f(qkv[base + Hc + d]),       k1 = b2f(qkv[base + Hc + d + 64]);
    float v0 = b2f(qkv[base + 2 * Hc + d]),   v1 = b2f(qkv[base + 2 * Hc + d + 64]);

    float c0 = cosp[s * HDc + d], c1 = cosp[s * HDc + d + 64];
    float s0 = sinp[s * HDc + d], s1 = sinp[s * HDc + d + 64];
    float ln = logn[s];

    float qo0 = (q0 * c0 - q1 * s0) * ln;
    float qo1 = (q1 * c1 + q0 * s1) * ln;
    float ko0 = (k0 * c0 - k1 * s0);
    float ko1 = (k1 * c1 + k0 * s1);

    const int bh = b * NHc + h;
    const size_t qk = ((size_t)bh * Sc + s) * HDc;
    qb[qk + d] = f2b(qo0);  qb[qk + d + 64] = f2b(qo1);
    kb[qk + d] = f2b(ko0);  kb[qk + d + 64] = f2b(ko1);
    vtb[((size_t)bh * HDc + d) * Sc + s]        = f2b(v0);
    vtb[((size_t)bh * HDc + d + 64) * Sc + s]   = f2b(v1);
}

// ---------------- kernel 5: flash attention (WMMA, online softmax) ----------------
// block = 128 threads (4 waves); each wave owns 16 query rows; block covers 64 rows.
__global__ __launch_bounds__(128)
void attn_kernel(const unsigned short* __restrict__ qb,
                 const unsigned short* __restrict__ kb,
                 const unsigned short* __restrict__ vtb,
                 unsigned short* __restrict__ ctx) {
    __shared__ unsigned short pbuf[4][16][32];

    const int nqblk = Sc / 64;
    const int qblk  = blockIdx.x % nqblk;
    const int bh    = blockIdx.x / nqblk;
    const int b = bh / NHc, h = bh % NHc;

    const int wave = threadIdx.x >> 5;
    const int lane = threadIdx.x & 31;
    const int n  = lane & 15;
    const int hi = lane >> 4;
    const int qrow0 = qblk * 64 + wave * 16;

    const unsigned short* qp  = qb  + (size_t)bh * Sc * HDc;
    const unsigned short* kp  = kb  + (size_t)bh * Sc * HDc;
    const unsigned short* vtp = vtb + (size_t)bh * HDc * Sc;

    // Q fragments: 16 rows x 128 K, as 4 A-fragments, kept resident
    AFrag qf[4];
    #pragma unroll
    for (int f = 0; f < 4; ++f) load_frag(qf[f], qp, HDc, qrow0, 32 * f, lane);

    v8f o[8];
    #pragma unroll
    for (int t = 0; t < 8; ++t) o[t] = (v8f)0.0f;
    float mi[8], li[8];
    #pragma unroll
    for (int r = 0; r < 8; ++r) { mi[r] = -1.0e30f; li[r] = 0.0f; }

    const int jmax = qrow0 + 16;            // causal: keys <= max row in this wave
    for (int j0 = 0; j0 < jmax; j0 += 32) {
        // prefetch next key tile's K rows
        if (j0 + 32 < jmax) {
            __builtin_prefetch(kp + (size_t)(j0 + 32 + n) * HDc, 0, 1);
            __builtin_prefetch(kp + (size_t)(j0 + 48 + n) * HDc, 0, 1);
        }
        // ---- scores for 32 keys: two 16x16 WMMA stacks over K=128 ----
        v8f s0 = (v8f)0.0f, s1 = (v8f)0.0f;
        #pragma unroll
        for (int f = 0; f < 4; ++f) {
            AFrag bk;
            load_frag(bk, kp, HDc, j0, 32 * f, lane);
            s0 = wmma_bf16(qf[f], bk, s0);
            load_frag(bk, kp, HDc, j0 + 16, 32 * f, lane);
            s1 = wmma_bf16(qf[f], bk, s1);
        }
        // ---- online softmax across this 32-key tile ----
        #pragma unroll
        for (int r = 0; r < 8; ++r) {
            int row = qrow0 + r + 8 * hi;
            float e0 = s0[r] * SCALEF + ((j0 + n)      <= row ? 0.0f : NEGF);
            float e1 = s1[r] * SCALEF + ((j0 + 16 + n) <= row ? 0.0f : NEGF);
            float rmax = fmaxf(e0, e1);
            #pragma unroll
            for (int off = 8; off >= 1; off >>= 1)
                rmax = fmaxf(rmax, __shfl_xor(rmax, off));
            float mnew  = fmaxf(mi[r], rmax);
            float alpha = __expf(mi[r] - mnew);
            float p0 = __expf(e0 - mnew);
            float p1 = __expf(e1 - mnew);
            float psum = p0 + p1;
            #pragma unroll
            for (int off = 8; off >= 1; off >>= 1)
                psum += __shfl_xor(psum, off);
            li[r] = li[r] * alpha + psum;
            mi[r] = mnew;
            #pragma unroll
            for (int t = 0; t < 8; ++t) o[t][r] *= alpha;
            pbuf[wave][r + 8 * hi][n]      = f2b(p0);
            pbuf[wave][r + 8 * hi][n + 16] = f2b(p1);
        }
        // per-wave LDS RAW: explicit split-counter wait (no divergent barrier needed)
        asm volatile("s_wait_dscnt 0x0" ::: "memory");
        // ---- P(16x32) @ V(32x128): V pre-transposed so B pairs are contiguous ----
        AFrag pa;
        load_frag(pa, &pbuf[wave][0][0], 32, 0, 0, lane);
        #pragma unroll
        for (int t = 0; t < 8; ++t) {
            AFrag bv;
            load_frag(bv, vtp, Sc, 16 * t, j0, lane);   // row0 = d0, kcol = key base
            o[t] = wmma_bf16(pa, bv, o[t]);
        }
    }

    // ---- epilogue: normalize, write ctx bf16 [B,S,H] ----
    #pragma unroll
    for (int r = 0; r < 8; ++r) {
        float inv = 1.0f / li[r];
        int row = qrow0 + r + 8 * hi;
        #pragma unroll
        for (int t = 0; t < 8; ++t) {
            size_t idx = ((size_t)(b * Sc + row)) * Hc + h * HDc + 16 * t + n;
            ctx[idx] = f2b(o[t][r] * inv);
        }
    }
}

// ---------------- host-side launch ----------------
extern "C" void kernel_launch(void* const* d_in, const int* in_sizes, int n_in,
                              void* d_out, int out_size, void* d_ws, size_t ws_size,
                              hipStream_t stream) {
    const float* hs   = (const float*)d_in[0];
    const float* cosp = (const float*)d_in[1];
    const float* sinp = (const float*)d_in[2];
    // d_in[3] = attention_mask (causal, computed analytically in-kernel)
    const float* logn = (const float*)d_in[4];
    const float* Wc   = (const float*)d_in[5];
    const float* bc   = (const float*)d_in[6];
    const float* Wp   = (const float*)d_in[7];
    const float* bp   = (const float*)d_in[8];
    float* out = (float*)d_out;

    // workspace carve-up (bf16 buffers), total ~160 MiB
    char* w = (char*)d_ws;
    unsigned short* hsb  = (unsigned short*)w;  w += (size_t)BSc * Hc * 2;        // 16 MiB
    unsigned short* Wct  = (unsigned short*)w;  w += (size_t)H3c * Hc * 2;        // 24 MiB
    unsigned short* Wpt  = (unsigned short*)w;  w += (size_t)Hc * Hc * 2;         //  8 MiB
    unsigned short* qkvb = (unsigned short*)w;  w += (size_t)BSc * H3c * 2;       // 48 MiB
    unsigned short* qb   = (unsigned short*)w;  w += (size_t)Bc * NHc * Sc * HDc * 2;
    unsigned short* kbuf = (unsigned short*)w;  w += (size_t)Bc * NHc * Sc * HDc * 2;
    unsigned short* vtb  = (unsigned short*)w;  w += (size_t)Bc * NHc * Sc * HDc * 2;
    unsigned short* ctxb = (unsigned short*)w;  w += (size_t)BSc * Hc * 2;        // 16 MiB

    // 1) activations -> bf16
    {
        int n = BSc * Hc;
        cvt_f2b_kernel<<<(n + 255) / 256, 256, 0, stream>>>(hs, hsb, n);
    }
    // 2) weights -> bf16, transposed to [N][K]
    {
        int n = Hc * H3c;
        transpose_f2b_kernel<<<(n + 255) / 256, 256, 0, stream>>>(Wc, Wct, Hc, H3c);
        n = Hc * Hc;
        transpose_f2b_kernel<<<(n + 255) / 256, 256, 0, stream>>>(Wp, Wpt, Hc, Hc);
    }
    // 3) qkv = hs @ Wc + bc   (4096 x 6144 x 2048, bf16 WMMA, bf16 out)
    gemm_wmma_kernel<true><<<dim3(H3c / 256, BSc / 64), 256, 0, stream>>>(
        hsb, Wct, bc, qkvb, BSc, H3c, Hc);
    // 4) RoPE + logn + head-major reshuffle (+ V transpose)
    {
        int n = Bc * NHc * Sc * (HDc / 2);
        rope_kernel<<<(n + 255) / 256, 256, 0, stream>>>(qkvb, cosp, sinp, logn,
                                                         qb, kbuf, vtb);
    }
    // 5) causal flash attention per (b,h)
    attn_kernel<<<Bc * NHc * (Sc / 64), 128, 0, stream>>>(qb, kbuf, vtb, ctxb);
    // 6) out = ctx @ Wp + bp   (4096 x 2048 x 2048, fp32 out)
    gemm_wmma_kernel<false><<<dim3(Hc / 256, BSc / 64), 256, 0, stream>>>(
        ctxb, Wpt, bp, out, BSc, Hc, Hc);

    (void)in_sizes; (void)n_in; (void)out_size; (void)ws_size;
}
